// VectorQuantizer_72911364817053
// MI455X (gfx1250) — matrix operand order repaired and checked by reading, not compiled, not run on hardware
//
#include <hip/hip_runtime.h>
#include <cstdint>
#include <cstddef>

// Problem shape (from reference): x (8,1024,256) -> ROWS=8192, D=256; codebook 8192x256.
#define ROWS  8192
#define DIM   256
#define CODES 8192
#define NCHUNKS 4

// Output tuple flat layout: quantized(2097152) | loss(1) | indices(8192) | perplexity(1) | usage(1)
#define OUT_Q_OFF     0
#define OUT_LOSS_OFF  2097152
#define OUT_IDX_OFF   2097153
#define OUT_PERP_OFF  2105345
#define OUT_USE_OFF   2105346

typedef __attribute__((ext_vector_type(16))) __bf16  v16bf;
typedef __attribute__((ext_vector_type(8)))  float   v8f;
typedef __attribute__((ext_vector_type(4)))  unsigned int u32x4;

union Frag { u32x4 q[2]; v16bf v; };   // 32 bytes = 16 bf16 = 8 VGPRs

// ---- CDNA5 async global->LDS staging (guarded; sync LDS copy fallback) ----
// Builtin signature (from clang diagnostic): (v4i global*, v4i local*, Imm, Imm).
// Cast through integers: AS1 shares the 64-bit flat representation for global;
// flat shared addresses map to the LDS offset by low-32-bit truncation (ISA 10.2),
// and AS3 pointers are 32-bit.
#if defined(__gfx1250__) && __has_builtin(__builtin_amdgcn_global_load_async_to_lds_b128) && __has_builtin(__builtin_amdgcn_s_wait_asynccnt)
#define VQ_ASYNC 1
typedef __attribute__((ext_vector_type(4))) int vq_v4i;
typedef vq_v4i __attribute__((address_space(1)))* vq_gptr_t;
typedef vq_v4i __attribute__((address_space(3)))* vq_lptr_t;
#define VQ_STAGE16(src, dst, off)                                           \
  __builtin_amdgcn_global_load_async_to_lds_b128(                           \
      (vq_gptr_t)(unsigned long long)(uintptr_t)(src),                      \
      (vq_lptr_t)(unsigned int)(uintptr_t)(dst), (off), 0)
#define VQ_ASYNC_WAIT(n) __builtin_amdgcn_s_wait_asynccnt(n)
#else
#define VQ_ASYNC 0
#define VQ_STAGE16(src, dst, off) \
  (*(u32x4*)((char*)(dst) + (off)) = *(const u32x4*)((const char*)(src) + (off)))
#define VQ_ASYNC_WAIT(n) ((void)0)
#endif

__device__ __forceinline__ unsigned short f2bf(float f) {
  union { float f; unsigned u; } v; v.f = f;
  unsigned r = v.u;
  r += 0x7fffu + ((r >> 16) & 1u);     // round-to-nearest-even
  return (unsigned short)(r >> 16);
}

// ---------------------------------------------------------------- init ws ----
__global__ void vq_init_ws(int* counts, float* sqerr) {
  int i = blockIdx.x * 256 + threadIdx.x;
  if (i < CODES) counts[i] = 0;
  if (i == 0)    *sqerr = 0.0f;
}

// ------------------------------------------------- row L2-normalize -> bf16 --
__global__ void vq_norm_rows(const float* __restrict__ src,
                             unsigned short* __restrict__ dst) {
  const int row = blockIdx.x;
  const int t   = threadIdx.x;          // 256 threads, one per element
  float v  = src[(size_t)row * DIM + t];
  float ss = v * v;
  #pragma unroll
  for (int off = 16; off; off >>= 1) ss += __shfl_xor(ss, off, 32);
  __shared__ float sp[8];
  if ((t & 31) == 0) sp[t >> 5] = ss;
  __syncthreads();
  float tot = 0.0f;
  #pragma unroll
  for (int i = 0; i < 8; ++i) tot += sp[i];
  float rs = rsqrtf(tot + 1e-12f);
  dst[(size_t)row * DIM + t] = f2bf(v * rs);
}

// ------------------------------------------------ WMMA argmax over codebook --
// Grid (64, NCHUNKS). Block: 256 threads (8 waves). Wave w owns the 16-row
// M-tile (blockIdx.x*128 + w*16); all waves share B tiles staged in LDS
// (double-buffered, async global->LDS). Each block sweeps CODES/NCHUNKS codes
// and writes per-chunk argmax partials.
__global__ void __launch_bounds__(256)
vq_argmax(const unsigned short* __restrict__ xn,
          const unsigned short* __restrict__ cbn,
          float* __restrict__ part_best,
          int* __restrict__ part_idx) {
  const int tid    = threadIdx.x;
  const int wave   = tid >> 5;
  const int lane   = tid & 31;
  const int col    = lane & 15;         // matrix column / row-within-tile
  const int khalf  = lane >> 4;         // 0: low K half, 1: high K half
  const int mTile  = blockIdx.x * 128 + wave * 16;
  const int nBeg   = blockIdx.y * (CODES / NCHUNKS);
  const int NTILES = (CODES / NCHUNKS) / 16;   // 128

  __shared__ __align__(16) char smem[2][16 * DIM * 2];   // 2 x 8 KiB B tiles

  // ---- preload A fragments for the whole D=256 (8 K-steps of 32) ----
  // 16-bit A layout: lanes 0-15 -> M=lane, K {k0..k0+7, k0+16..k0+23}
  //                  lanes 16-31 -> M=lane-16, K {k0+8..k0+15, k0+24..k0+31}
  Frag a[8];
  {
    const unsigned short* arow = xn + (size_t)(mTile + col) * DIM;
    #pragma unroll
    for (int ks = 0; ks < 8; ++ks) {
      const int k0 = ks * 32;
      a[ks].q[0] = *(const u32x4*)(arow + k0 +      khalf * 8);
      a[ks].q[1] = *(const u32x4*)(arow + k0 + 16 + khalf * 8);
    }
  }

  float best[8];
  int   bidx[8];
  #pragma unroll
  for (int r = 0; r < 8; ++r) { best[r] = -3.0e38f; bidx[r] = 0; }

  const char* cbB = (const char*)(cbn + (size_t)nBeg * DIM);   // chunk base

  // stage tile 0 (each thread moves 32B of the 8 KiB tile)
  {
    const char* src = cbB + tid * 32;
    char*       dst = &smem[0][tid * 32];
    VQ_STAGE16(src, dst, 0);
    VQ_STAGE16(src, dst, 16);
  }

  for (int i = 0; i < NTILES; ++i) {
    if (i + 1 < NTILES) {               // prefetch next tile into other buffer
      const char* src = cbB + (size_t)(i + 1) * (16 * DIM * 2) + tid * 32;
      char*       dst = &smem[(i + 1) & 1][tid * 32];
      VQ_STAGE16(src, dst, 0);
      VQ_STAGE16(src, dst, 16);
      VQ_ASYNC_WAIT(2);                 // tile i done; tile i+1 in flight
    } else {
      VQ_ASYNC_WAIT(0);
    }
    __syncthreads();                    // all waves' stage of tile i visible

    // B layout (32x16 bf16): lane = column; lanes 0-15 hold K {0..7, 16..23},
    // lanes 16-31 hold K {8..15, 24..31} (matches A).
    const char* bufp = smem[i & 1] + col * (DIM * 2) + khalf * 16;
    v8f c = {};
    #pragma unroll
    for (int ks = 0; ks < 8; ++ks) {
      Frag b;
      const u32x4* p = (const u32x4*)(bufp + ks * 64);
      b.q[0] = p[0];
      b.q[1] = p[2];
      c = __builtin_amdgcn_wmma_f32_16x16x32_bf16(
              false, a[ks].v, false, b.v, (short)0, c, false, false);
    }

    // C/D layout: VGPR r -> row mTile+r (lanes 0-15) or mTile+8+r (lanes 16-31)
    const int myN = nBeg + i * 16 + col;
    #pragma unroll
    for (int r = 0; r < 8; ++r) {
      if (c[r] > best[r]) { best[r] = c[r]; bidx[r] = myN; }
    }
    __syncthreads();                    // reads done before buffer is re-staged
  }

  // ---- argmax across the 16 columns held by each half-wave ----
  #pragma unroll
  for (int r = 0; r < 8; ++r) {
    #pragma unroll
    for (int off = 8; off; off >>= 1) {
      float os = __shfl_xor(best[r], off, 32);
      int   oi = __shfl_xor(bidx[r], off, 32);
      if (os > best[r] || (os == best[r] && oi < bidx[r])) {
        best[r] = os; bidx[r] = oi;
      }
    }
  }

  // each wave owns its rows exclusively -> store partials directly
  if (col == 0) {                       // lanes 0 and 16
    float* pb = part_best + (size_t)blockIdx.y * ROWS + mTile + khalf * 8;
    int*   pi = part_idx  + (size_t)blockIdx.y * ROWS + mTile + khalf * 8;
    #pragma unroll
    for (int r = 0; r < 8; ++r) { pb[r] = best[r]; pi[r] = bidx[r]; }
  }
}

// ----------------------------------------- combine per-chunk argmax partials --
__global__ void vq_reduce_parts(const float* __restrict__ part_best,
                                const int* __restrict__ part_idx,
                                int* __restrict__ out_idx) {
  const int r = blockIdx.x * 256 + threadIdx.x;
  float bs = part_best[r];
  int   bi = part_idx[r];
  #pragma unroll
  for (int c = 1; c < NCHUNKS; ++c) {
    float os = part_best[(size_t)c * ROWS + r];
    int   oi = part_idx [(size_t)c * ROWS + r];
    if (os > bs || (os == bs && oi < bi)) { bs = os; bi = oi; }
  }
  out_idx[r] = bi;
}

// -------------------------------------------- gather + histogram + sq-error --
__global__ void vq_gather_stats(const float* __restrict__ x,
                                const float* __restrict__ cb,
                                const int* __restrict__ idx,
                                float* __restrict__ out_q,
                                float* __restrict__ out_idx,
                                int* __restrict__ counts,
                                float* __restrict__ sqerr) {
  const int row = blockIdx.x;
  const int t   = threadIdx.x;
  const int k   = idx[row];
  float q  = cb[(size_t)k * DIM + t];
  float xv = x[(size_t)row * DIM + t];
  out_q[(size_t)row * DIM + t] = q;     // STE output == quantized value
  float d  = q - xv;
  float ss = d * d;
  #pragma unroll
  for (int off = 16; off; off >>= 1) ss += __shfl_xor(ss, off, 32);
  __shared__ float sp[8];
  if ((t & 31) == 0) sp[t >> 5] = ss;
  __syncthreads();
  if (t == 0) {
    float tot = 0.0f;
    #pragma unroll
    for (int i = 0; i < 8; ++i) tot += sp[i];
    atomicAdd(sqerr, tot);
    atomicAdd(&counts[k], 1);
    out_idx[row] = (float)k;
  }
}

// ---------------------------------------------------------- scalar finalize --
__global__ void vq_finalize(const int* __restrict__ counts,
                            const float* __restrict__ sqerr,
                            float* __restrict__ out) {
  const int t = threadIdx.x;            // 1024 threads
  float nplogp = 0.0f;
  int   used   = 0;
  for (int k = t; k < CODES; k += 1024) {
    int c = counts[k];
    float p = (float)c * (1.0f / (float)ROWS);
    nplogp += -p * logf(p + 1e-10f);
    used   += (c > 0) ? 1 : 0;
  }
  #pragma unroll
  for (int off = 16; off; off >>= 1) {
    nplogp += __shfl_xor(nplogp, off, 32);
    used   += __shfl_xor(used,   off, 32);
  }
  __shared__ float sf[32];
  __shared__ int   si[32];
  if ((t & 31) == 0) { sf[t >> 5] = nplogp; si[t >> 5] = used; }
  __syncthreads();
  if (t == 0) {
    float lp = 0.0f; int u = 0;
    #pragma unroll
    for (int i = 0; i < 32; ++i) { lp += sf[i]; u += si[i]; }
    float mse = *sqerr * (1.0f / ((float)ROWS * (float)DIM));
    // 2*(0.25*mse + mse) + 0.1*log_perplexity
    out[OUT_LOSS_OFF] = 2.5f * mse + 0.1f * lp;
    out[OUT_PERP_OFF] = expf(lp);
    out[OUT_USE_OFF]  = (float)u * (1.0f / (float)CODES);
  }
}

// ---------------------------------------------------------------------------
extern "C" void kernel_launch(void* const* d_in, const int* in_sizes, int n_in,
                              void* d_out, int out_size, void* d_ws, size_t ws_size,
                              hipStream_t stream) {
  const float* x  = (const float*)d_in[0];   // (8,1024,256) f32
  const float* cb = (const float*)d_in[1];   // (8192,256)  f32
  float* out = (float*)d_out;

  char* ws = (char*)d_ws;
  unsigned short* xn  = (unsigned short*)(ws);                 // 4 MiB bf16 Xn
  unsigned short* cbn = (unsigned short*)(ws + 4194304);       // 4 MiB bf16 Cn
  float* part_best = (float*)(ws + 8388608);                   // 128 KiB
  int*   part_idx  = (int*)  (ws + 8519680);                   // 128 KiB
  int*   idx       = (int*)  (ws + 8650752);                   // 32 KiB
  int*   counts    = (int*)  (ws + 8683520);                   // 32 KiB
  float* sqerr     = (float*)(ws + 8716288);                   // 4 B

  vq_init_ws   <<<(CODES + 255) / 256, 256, 0, stream>>>(counts, sqerr);
  vq_norm_rows <<<ROWS,  256, 0, stream>>>(x,  xn);
  vq_norm_rows <<<CODES, 256, 0, stream>>>(cb, cbn);

  dim3 gridA(ROWS / 128, NCHUNKS);
  vq_argmax    <<<gridA, 256, 0, stream>>>(xn, cbn, part_best, part_idx);
  vq_reduce_parts<<<ROWS / 256, 256, 0, stream>>>(part_best, part_idx, idx);

  vq_gather_stats<<<ROWS, 256, 0, stream>>>(x, cb, idx,
                                            out + OUT_Q_OFF, out + OUT_IDX_OFF,
                                            counts, sqerr);
  vq_finalize  <<<1, 1024, 0, stream>>>(counts, sqerr, out);
}
